// Model_59365037965465
// MI455X (gfx1250) — matrix operand order repaired and checked by reading, not compiled, not run on hardware
//
#include <hip/hip_runtime.h>
#include <hip/hip_bf16.h>

// ---------------------------------------------------------------------------
// Types for CDNA5 WMMA
// ---------------------------------------------------------------------------
typedef __bf16 bf16_t;
typedef bf16_t v16bf __attribute__((ext_vector_type(16)));
typedef bf16_t v8bf  __attribute__((ext_vector_type(8)));
typedef float  v8f   __attribute__((ext_vector_type(8)));

#define TM 128
#define TN 64
#define TK 32

// ---------------------------------------------------------------------------
// Generic C = act(A@W [+ Cprev] [+ bias]) GEMM, bf16 WMMA, f32 accumulate.
// A: MxK fp32 row-major (lda), W: KxN fp32 row-major, C: fp32 (ldc).
// REQUIREMENT: N must be a multiple of TN (true for all launches: 128/256).
// Block = 256 threads = 8 waves in a 4x2 grid; each wave owns a 32x32 patch
// (2 M-tiles x 2 N-tiles -> 4 v_wmma per K step).
// ---------------------------------------------------------------------------
__global__ __launch_bounds__(256)
void gemm_bf16_wmma(const float* __restrict__ A, int lda, int M, int K,
                    const float* __restrict__ W, int N,
                    const float* __restrict__ bias,
                    float* __restrict__ C, int ldc,
                    int accumulate, int do_relu)
{
    __shared__ __align__(16) bf16_t As[TM][TK + 8];
    __shared__ __align__(16) bf16_t Bs[TN][TK + 8];

    const int tid  = threadIdx.x;
    const int lane = tid & 31;
    const int wv   = tid >> 5;          // 0..7 waves
    const int wrow = (wv & 3) * 32;     // wave row offset in tile (4 row-waves)
    const int wcol = (wv >> 2) * 32;    // wave col offset in tile (2 col-waves)
    const int half = lane >> 4;         // 0 or 1
    const int l16  = lane & 15;

    const int m0 = blockIdx.x * TM;
    const int n0 = blockIdx.y * TN;

    v8f acc00 = {}, acc01 = {}, acc10 = {}, acc11 = {};
    if (accumulate) {
        #pragma unroll
        for (int r = 0; r < 8; ++r) {
            int row0 = m0 + wrow + half * 8 + r;        // mt = 0
            int row1 = row0 + 16;                       // mt = 1
            int c0   = n0 + wcol + l16;                 // nt = 0
            int c1   = c0 + 16;                         // nt = 1
            if (row0 < M) { acc00[r] = C[(long)row0 * ldc + c0];
                            acc01[r] = C[(long)row0 * ldc + c1]; }
            if (row1 < M) { acc10[r] = C[(long)row1 * ldc + c0];
                            acc11[r] = C[(long)row1 * ldc + c1]; }
        }
    }

    // Staging thread mappings (computed once)
    const int a_row = tid >> 1;               // 0..127
    const int a_kb  = (tid & 1) * 16;         // 0 or 16
    const int b_nn  = tid & 63;               // 0..63
    const int b_kb  = (tid >> 6) * 8;         // 0,8,16,24

    for (int kt = 0; kt < K; kt += TK) {
        if (kt + TK <= K) {
            // ---------- fast path: no per-element guards ----------
            // A tile: thread loads 16 consecutive k of one row (clamped row,
            // select-to-zero; loads always legal, no exec predication).
            {
                int grow = m0 + a_row;
                int gr   = grow < M ? grow : (M - 1);
                const float* ap = A + (long)gr * lda + kt + a_kb;
                bool ok = grow < M;
                v8bf p0, p1;
                #pragma unroll
                for (int j = 0; j < 8; ++j) {
                    float f0 = ap[j];
                    float f1 = ap[8 + j];
                    p0[j] = (bf16_t)(ok ? f0 : 0.0f);
                    p1[j] = (bf16_t)(ok ? f1 : 0.0f);
                }
                *(v8bf*)&As[a_row][a_kb]     = p0;
                *(v8bf*)&As[a_row][a_kb + 8] = p1;
            }
            // B tile: thread gathers 8 consecutive k for one column n
            // (coalesced across threads per k row), one b128 LDS store.
            {
                const float* wp = W + (long)(kt + b_kb) * N + n0 + b_nn;
                v8bf q;
                #pragma unroll
                for (int j = 0; j < 8; ++j)
                    q[j] = (bf16_t)wp[(long)j * N];
                *(v8bf*)&Bs[b_nn][b_kb] = q;
            }
        } else {
            // ---------- slow path: partial K tile (fc1, K=259 tail) ----------
            {
                int grow = m0 + a_row;
                bool rok = grow < M;
                const float* ap = A + (long)(rok ? grow : 0) * lda;
                v8bf p0, p1;
                #pragma unroll
                for (int j = 0; j < 8; ++j) {
                    int k0 = kt + a_kb + j;
                    int k1 = k0 + 8;
                    p0[j] = (bf16_t)((rok && k0 < K) ? ap[k0] : 0.0f);
                    p1[j] = (bf16_t)((rok && k1 < K) ? ap[k1] : 0.0f);
                }
                *(v8bf*)&As[a_row][a_kb]     = p0;
                *(v8bf*)&As[a_row][a_kb + 8] = p1;
            }
            {
                v8bf q;
                #pragma unroll
                for (int j = 0; j < 8; ++j) {
                    int gk = kt + b_kb + j;
                    q[j] = (bf16_t)((gk < K) ? W[(long)gk * N + n0 + b_nn] : 0.0f);
                }
                *(v8bf*)&Bs[b_nn][b_kb] = q;
            }
        }
        __syncthreads();

        // Fragments per ISA 7.12.2 layouts.
        // A: lane = row l16 (+16 for second m-tile); elems 0..7: K=half*8+j,
        //    elems 8..15: K=16+half*8+j  -> two 16B LDS reads each.
        int am0 = wrow + l16;
        v8bf a0lo = *(const v8bf*)&As[am0][half * 8];
        v8bf a0hi = *(const v8bf*)&As[am0][16 + half * 8];
        v16bf afrag0 = __builtin_shufflevector(a0lo, a0hi,
            0,1,2,3,4,5,6,7,8,9,10,11,12,13,14,15);
        int am1 = am0 + 16;
        v8bf a1lo = *(const v8bf*)&As[am1][half * 8];
        v8bf a1hi = *(const v8bf*)&As[am1][16 + half * 8];
        v16bf afrag1 = __builtin_shufflevector(a1lo, a1hi,
            0,1,2,3,4,5,6,7,8,9,10,11,12,13,14,15);

        // B: lane = col l16 (+16 for second n-tile); elems 0..7: K=half*8+j,
        //    elems 8..15: K=16+half*8+j (same as A).
        int bn0 = wcol + l16;
        v8bf b0lo = *(const v8bf*)&Bs[bn0][half * 8];
        v8bf b0hi = *(const v8bf*)&Bs[bn0][16 + half * 8];
        v16bf bfrag0 = __builtin_shufflevector(b0lo, b0hi,
            0,1,2,3,4,5,6,7,8,9,10,11,12,13,14,15);
        int bn1 = bn0 + 16;
        v8bf b1lo = *(const v8bf*)&Bs[bn1][half * 8];
        v8bf b1hi = *(const v8bf*)&Bs[bn1][16 + half * 8];
        v16bf bfrag1 = __builtin_shufflevector(b1lo, b1hi,
            0,1,2,3,4,5,6,7,8,9,10,11,12,13,14,15);

        acc00 = __builtin_amdgcn_wmma_f32_16x16x32_bf16(
                    false, afrag0, false, bfrag0, (short)0, acc00, false, false);
        acc01 = __builtin_amdgcn_wmma_f32_16x16x32_bf16(
                    false, afrag0, false, bfrag1, (short)0, acc01, false, false);
        acc10 = __builtin_amdgcn_wmma_f32_16x16x32_bf16(
                    false, afrag1, false, bfrag0, (short)0, acc10, false, false);
        acc11 = __builtin_amdgcn_wmma_f32_16x16x32_bf16(
                    false, afrag1, false, bfrag1, (short)0, acc11, false, false);
        __syncthreads();
    }

    // Store (N is a multiple of TN so columns always in range).
    #pragma unroll
    for (int r = 0; r < 8; ++r) {
        int row0 = m0 + wrow + half * 8 + r;
        int row1 = row0 + 16;
        int c0   = n0 + wcol + l16;
        int c1   = c0 + 16;
        float b0v = bias ? bias[c0] : 0.0f;
        float b1v = bias ? bias[c1] : 0.0f;
        if (row0 < M) {
            float v0 = acc00[r] + b0v, v1 = acc01[r] + b1v;
            if (do_relu) { v0 = fmaxf(v0, 0.0f); v1 = fmaxf(v1, 0.0f); }
            C[(long)row0 * ldc + c0] = v0;
            C[(long)row0 * ldc + c1] = v1;
        }
        if (row1 < M) {
            float v0 = acc10[r] + b0v, v1 = acc11[r] + b1v;
            if (do_relu) { v0 = fmaxf(v0, 0.0f); v1 = fmaxf(v1, 0.0f); }
            C[(long)row1 * ldc + c0] = v0;
            C[(long)row1 * ldc + c1] = v1;
        }
    }
}

// ---------------------------------------------------------------------------
// Helpers / elementwise kernels
// ---------------------------------------------------------------------------
__device__ __forceinline__ void atomicMaxF(float* addr, float val) {
    unsigned int* ua = (unsigned int*)addr;
    unsigned int old = *ua;
    while (__uint_as_float(old) < val) {
        unsigned int assumed = old;
        old = atomicCAS(ua, assumed, __float_as_uint(val));
        if (old == assumed) break;
    }
}

__global__ void fill_f32(float* __restrict__ p, float v, long n) {
    long i = (long)blockIdx.x * blockDim.x + threadIdx.x;
    if (i < n) p[i] = v;
}

// GAT pass 1: per (edge, h) leaky-relu attention logit -> segment max
__global__ void gat_max_k(const float* __restrict__ x,
                          const int* __restrict__ src, const int* __restrict__ dst,
                          long E,
                          const float* __restrict__ gw, const float* __restrict__ asv,
                          const float* __restrict__ adv,
                          float* __restrict__ amax)
{
    long idx = (long)blockIdx.x * blockDim.x + threadIdx.x;
    if (idx >= E * 256) return;
    int e = (int)(idx >> 8), h = (int)(idx & 255);
    int s = src[e], d = dst[e];
    float g  = gw[h];
    float a  = x[(long)s * 19] * g * asv[h] + x[(long)d * 19] * g * adv[h];
    a = a > 0.0f ? a : 0.2f * a;
    atomicMaxF(&amax[(long)d * 256 + h], a);
}

// GAT pass 2: exp(alpha - max), accumulate denominator and weighted numerator
__global__ void gat_sum_k(const float* __restrict__ x,
                          const int* __restrict__ src, const int* __restrict__ dst,
                          long E,
                          const float* __restrict__ gw, const float* __restrict__ asv,
                          const float* __restrict__ adv,
                          const float* __restrict__ amax,
                          float* __restrict__ den, float* __restrict__ num)
{
    long idx = (long)blockIdx.x * blockDim.x + threadIdx.x;
    if (idx >= E * 256) return;
    int e = (int)(idx >> 8), h = (int)(idx & 255);
    int s = src[e], d = dst[e];
    float g  = gw[h];
    float ps = x[(long)s * 19];
    float a  = ps * g * asv[h] + x[(long)d * 19] * g * adv[h];
    a = a > 0.0f ? a : 0.2f * a;
    long o  = (long)d * 256 + h;
    float ex = __expf(a - amax[o]);
    atomicAdd(&den[o], ex);
    atomicAdd(&num[o], ex * ps * g);     // xl[src][h] = price[src]*g_w[h]
}

// GAT finalize: p[i] = relu(mean_h(num/(den+eps)) + b)  (wave per node)
__global__ void gat_fin_k(const float* __restrict__ num, const float* __restrict__ den,
                          const float* __restrict__ b, float* __restrict__ p, int n)
{
    int node = blockIdx.x * 8 + (threadIdx.x >> 5);
    int lane = threadIdx.x & 31;
    if (node >= n) return;
    float s = 0.0f;
    long base = (long)node * 256;
    for (int h = lane; h < 256; h += 32)
        s += num[base + h] / (den[base + h] + 1e-16f);
    for (int off = 16; off; off >>= 1) s += __shfl_xor(s, off, 32);
    if (lane == 0) p[node] = fmaxf(s * (1.0f / 256.0f) + b[0], 0.0f);
}

// x1 = concat([x3, p1, p2, p3], axis=1)  (259 cols)
__global__ void build_x1_k(const float* __restrict__ x3,
                           const float* __restrict__ p1, const float* __restrict__ p2,
                           const float* __restrict__ p3,
                           float* __restrict__ x1, int n)
{
    long idx = (long)blockIdx.x * blockDim.x + threadIdx.x;
    if (idx >= (long)n * 259) return;
    int i = (int)(idx / 259), c = (int)(idx % 259);
    float v;
    if (c < 256)       v = x3[(long)i * 256 + c];
    else if (c == 256) v = p1[i];
    else if (c == 257) v = p2[i];
    else               v = p3[i];
    x1[idx] = v;
}

// SAGE scatter: agg[dst] += x1[src], cnt[dst] += 1
__global__ void sage_agg_k(const float* __restrict__ x1,
                           const int* __restrict__ src, const int* __restrict__ dst,
                           long E, float* __restrict__ agg, float* __restrict__ cnt)
{
    long idx = (long)blockIdx.x * blockDim.x + threadIdx.x;
    if (idx >= E * 259) return;
    int e = (int)(idx / 259), c = (int)(idx % 259);
    int s = src[e], d = dst[e];
    atomicAdd(&agg[(long)d * 259 + c], x1[(long)s * 259 + c]);
    if (c == 0) atomicAdd(&cnt[d], 1.0f);
}

__global__ void sage_mean_k(float* __restrict__ agg, const float* __restrict__ cnt, int n)
{
    long idx = (long)blockIdx.x * blockDim.x + threadIdx.x;
    if (idx >= (long)n * 259) return;
    agg[idx] /= fmaxf(cnt[idx / 259], 1.0f);
}

// out[i] = A[i,:] . w + b   (wave per row)
__global__ void rowdot_k(const float* __restrict__ A, int K,
                         const float* __restrict__ w, const float* __restrict__ b,
                         float* __restrict__ out, int n)
{
    int row  = blockIdx.x * 8 + (threadIdx.x >> 5);
    int lane = threadIdx.x & 31;
    if (row >= n) return;
    float s = 0.0f;
    for (int k = lane; k < K; k += 32) s += A[(long)row * K + k] * w[k];
    for (int off = 16; off; off >>= 1) s += __shfl_xor(s, off, 32);
    if (lane == 0) out[row] = s + b[0];
}

// ---------------------------------------------------------------------------
// Host orchestration
// ---------------------------------------------------------------------------
static inline void launch_gemm(const float* A, int lda, int M, int K,
                               const float* W, int N, const float* bias,
                               float* C, int ldc, int acc, int relu, hipStream_t s)
{
    dim3 grid((M + TM - 1) / TM, (N + TN - 1) / TN), blk(256);
    gemm_bf16_wmma<<<grid, blk, 0, s>>>(A, lda, M, K, W, N, bias, C, ldc, acc, relu);
}

static inline unsigned int nblk(long total, int t) { return (unsigned int)((total + t - 1) / t); }

extern "C" void kernel_launch(void* const* d_in, const int* in_sizes, int n_in,
                              void* d_out, int out_size, void* d_ws, size_t ws_size,
                              hipStream_t stream)
{
    const int   N    = in_sizes[0] / 19;         // 50000
    const long  E12  = in_sizes[37] / 2;         // 250000
    const long  E3   = in_sizes[39] / 2;         // 500000

    const float* x      = (const float*)d_in[0];
    const float* fc1_w  = (const float*)d_in[1];  const float* fc1_b = (const float*)d_in[2];
    const float* fc2_w  = (const float*)d_in[3];  const float* fc2_b = (const float*)d_in[4];
    const float* fc3_w  = (const float*)d_in[5];  const float* fc3_b = (const float*)d_in[6];
    const float* g_w[3]  = {(const float*)d_in[7],  (const float*)d_in[11], (const float*)d_in[15]};
    const float* g_as[3] = {(const float*)d_in[8],  (const float*)d_in[12], (const float*)d_in[16]};
    const float* g_ad[3] = {(const float*)d_in[9],  (const float*)d_in[13], (const float*)d_in[17]};
    const float* g_b[3]  = {(const float*)d_in[10], (const float*)d_in[14], (const float*)d_in[18]};
    const float* sg_wl[3] = {(const float*)d_in[19], (const float*)d_in[21], (const float*)d_in[23]};
    const float* sg_wr[3] = {(const float*)d_in[20], (const float*)d_in[22], (const float*)d_in[24]};
    const float* c21_wl = (const float*)d_in[25]; const float* c21_wr = (const float*)d_in[26];
    const float* fc4_w  = (const float*)d_in[27]; const float* fc4_b  = (const float*)d_in[28];
    const float* fc4n_w = (const float*)d_in[29]; const float* fc4n_b = (const float*)d_in[30];
    const float* fc5_w  = (const float*)d_in[31]; const float* fc5_b  = (const float*)d_in[32];
    const float* fc6_w  = (const float*)d_in[33]; const float* fc6_b  = (const float*)d_in[34];
    const float* fc7_w  = (const float*)d_in[35]; const float* fc7_b  = (const float*)d_in[36];
    const int* r1  = (const int*)d_in[37];
    const int* r2  = (const int*)d_in[38];
    const int* r3  = (const int*)d_in[39];
    const int* neg = (const int*)d_in[40];

    // Workspace layout (fp32), with reuse: A1/A2/A3 are GAT scratch then y/y2/xn1
    float* ws = (float*)d_ws;
    float* A0  = ws;                 // N*256 : x3 -> later xn2
    float* A1  = A0 + (long)N * 256; // N*256 : amax -> y
    float* A2  = A1 + (long)N * 256; // N*256 : den  -> y2
    float* A3  = A2 + (long)N * 256; // N*256 : num  -> xn1
    float* X1  = A3 + (long)N * 256; // N*259
    float* AGG = X1 + (long)N * 259; // N*259
    float* S   = AGG + (long)N * 259;// N*384
    float* P1  = S + (long)N * 384;  // N
    float* P2  = P1 + N;
    float* P3  = P2 + N;
    float* CNT = P3 + N;
    float* yout  = (float*)d_out;        // N
    float* xnout = yout + N;             // N

    // ---- MLP trunk: h=relu(x[:,1:]@fc1+b); h2=relu(h@fc2+b); x3=relu(h2@fc3+b)
    launch_gemm(x + 1, 19, N, 18, fc1_w, 256, fc1_b, A1, 256, 0, 1, stream);
    launch_gemm(A1, 256, N, 256, fc2_w, 256, fc2_b, A2, 256, 0, 1, stream);
    launch_gemm(A2, 256, N, 256, fc3_w, 256, fc3_b, A0, 256, 0, 1, stream);

    // ---- Three GAT heads on x_price
    const int*  edges[3] = {r1, r2, r3};
    const long  ecnt[3]  = {E12, E12, E3};
    float*      pout[3]  = {P1, P2, P3};
    for (int g = 0; g < 3; ++g) {
        long E = ecnt[g];
        const int* esrc = edges[g];
        const int* edst = edges[g] + E;
        fill_f32<<<nblk((long)N * 256, 256), 256, 0, stream>>>(A1, -3.0e38f, (long)N * 256);
        fill_f32<<<nblk((long)N * 256, 256), 256, 0, stream>>>(A2, 0.0f, (long)N * 256);
        fill_f32<<<nblk((long)N * 256, 256), 256, 0, stream>>>(A3, 0.0f, (long)N * 256);
        gat_max_k<<<nblk(E * 256, 256), 256, 0, stream>>>(x, esrc, edst, E,
                                                          g_w[g], g_as[g], g_ad[g], A1);
        gat_sum_k<<<nblk(E * 256, 256), 256, 0, stream>>>(x, esrc, edst, E,
                                                          g_w[g], g_as[g], g_ad[g], A1, A2, A3);
        gat_fin_k<<<nblk(N, 8), 256, 0, stream>>>(A3, A2, g_b[g], pout[g], N);
    }

    // ---- x1 = concat(x3, p1, p2, p3)
    build_x1_k<<<nblk((long)N * 259, 256), 256, 0, stream>>>(A0, P1, P2, P3, X1, N);

    // ---- SAGE aggregate over r3 (shared by sg1..3)
    fill_f32<<<nblk((long)N * 259, 256), 256, 0, stream>>>(AGG, 0.0f, (long)N * 259);
    fill_f32<<<nblk(N, 256), 256, 0, stream>>>(CNT, 0.0f, N);
    sage_agg_k<<<nblk(E3 * 259, 256), 256, 0, stream>>>(X1, r3, r3 + E3, E3, AGG, CNT);
    sage_mean_k<<<nblk((long)N * 259, 256), 256, 0, stream>>>(AGG, CNT, N);

    // s[:, seg*128:+128] = relu(mean@Wl + x1@Wr)
    for (int seg = 0; seg < 3; ++seg) {
        float* Cs = S + seg * 128;
        launch_gemm(AGG, 259, N, 259, sg_wl[seg], 128, nullptr, Cs, 384, 0, 0, stream);
        launch_gemm(X1,  259, N, 259, sg_wr[seg], 128, nullptr, Cs, 384, 1, 1, stream);
    }

    // ---- y head: fc4 -> fc5 -> fc6
    launch_gemm(S,  384, N, 384, fc4_w, 256, fc4_b, A1, 256, 0, 1, stream);
    launch_gemm(A1, 256, N, 256, fc5_w, 256, fc5_b, A2, 256, 0, 1, stream);
    rowdot_k<<<nblk(N, 8), 256, 0, stream>>>(A2, 256, fc6_w, fc6_b, yout, N);

    // ---- neg branch: SAGE over neg edges -> c21 -> fc4n -> fc7
    fill_f32<<<nblk((long)N * 259, 256), 256, 0, stream>>>(AGG, 0.0f, (long)N * 259);
    fill_f32<<<nblk(N, 256), 256, 0, stream>>>(CNT, 0.0f, N);
    sage_agg_k<<<nblk(E3 * 259, 256), 256, 0, stream>>>(X1, neg, neg + E3, E3, AGG, CNT);
    sage_mean_k<<<nblk((long)N * 259, 256), 256, 0, stream>>>(AGG, CNT, N);

    launch_gemm(AGG, 259, N, 259, c21_wl, 256, nullptr, A3, 256, 0, 0, stream);
    launch_gemm(X1,  259, N, 259, c21_wr, 256, nullptr, A3, 256, 1, 1, stream);
    launch_gemm(A3,  256, N, 256, fc4n_w, 256, fc4n_b, A0, 256, 0, 1, stream);
    rowdot_k<<<nblk(N, 8), 256, 0, stream>>>(A0, 256, fc7_w, fc7_b, xnout, N);
}
